// WindowAttentionV2_37469294690493
// MI455X (gfx1250) — compile-verified, matrix operand order checked by reading, not run on hardware
//
#include <hip/hip_runtime.h>
#include <hip/hip_bf16.h>
#include <math.h>

// ---------------------------------------------------------------------------
// Swin-V2 window attention for MI455X (gfx1250, wave32, WMMA).
// bf16 WMMA (v_wmma_f32_16x16x32_bf16) with f32 accumulation; fused per-window
// attention so the (B,H,N,N) tensor never touches HBM. Memory-bound target:
// ~0.6 GB min traffic -> ~26 us at 23.3 TB/s. CDNA5 data movement:
// GLOBAL_LOAD_ASYNC_TO_LDS (ASYNCcnt) for the x-window stage and a TDM
// (tensor_load_to_lds, TENSORcnt) descriptor for the proj-GEMM A tiles.
// ---------------------------------------------------------------------------

typedef __bf16 bf16;
typedef __attribute__((ext_vector_type(16))) __bf16 v16bf;
typedef __attribute__((ext_vector_type(8)))  float  v8f;
typedef __attribute__((ext_vector_type(4)))  unsigned u32x4;
typedef __attribute__((ext_vector_type(8)))  int      i32x8;
typedef __attribute__((ext_vector_type(4)))  int      i32x4;

typedef __attribute__((address_space(1))) i32x4 as1_i32x4;  // global v4i
typedef __attribute__((address_space(3))) i32x4 as3_i32x4;  // LDS v4i
typedef __attribute__((address_space(3))) void  as3_void;

#if defined(__has_builtin)
#if __has_builtin(__builtin_amdgcn_global_load_async_to_lds_b128) && \
    __has_builtin(__builtin_amdgcn_s_wait_asynccnt)
#define USE_ASYNC_LDS 1
#endif
#if __has_builtin(__builtin_amdgcn_tensor_load_to_lds) && \
    __has_builtin(__builtin_amdgcn_s_wait_tensorcnt)
#define USE_TDM 1
#endif
#endif
#ifndef USE_ASYNC_LDS
#define USE_ASYNC_LDS 0
#endif
#ifndef USE_TDM
#define USE_TDM 0
#endif

struct alignas(16) bf16x8 { bf16 x[8]; };

#if USE_ASYNC_LDS
// 16B global -> LDS async copy. Builtin signature (probe-confirmed via
// diagnostics): (v4i addrspace(1)*, v4i addrspace(3)*, imm offset, imm cpol).
// Integer casts: flat global VA -> AS1 ptr; low 32 bits of a flat LDS pointer
// are the LDS byte offset -> AS3 ptr.
__device__ __forceinline__ void async_copy16(const void* gsrc, void* ldst) {
  __builtin_amdgcn_global_load_async_to_lds_b128(
      (as1_i32x4*)(unsigned long long)gsrc,
      (as3_i32x4*)(unsigned)(unsigned long long)ldst, 0, 0);
}
#endif

__device__ __forceinline__ bf16 f2bf(float f) {
  unsigned u = __builtin_bit_cast(unsigned, f);
  u += 0x7FFFu + ((u >> 16) & 1u);               // round-to-nearest-even
  return __builtin_bit_cast(bf16, (unsigned short)(u >> 16));
}

__device__ __forceinline__ unsigned pack2bf(float a, float b) {
  unsigned lo = (unsigned)__builtin_bit_cast(unsigned short, f2bf(a));
  unsigned hi = (unsigned)__builtin_bit_cast(unsigned short, f2bf(b));
  return lo | (hi << 16);
}

__device__ __forceinline__ v8f vzero8() {
  v8f z = {0.f,0.f,0.f,0.f,0.f,0.f,0.f,0.f};
  return z;
}

// 16-bit A/B fragment (16x32 A or 32x16 B) loader, row-major source.
// Per the CDNA5 ISA layout: lane L covers row (L&15); its 16 bf16 elements are
// k in [k0,k0+8) and [k0+16,k0+24) with k0 = 8*(L>>4): two 16-byte chunks.
__device__ __forceinline__ v16bf frag_ld(const bf16* base, int stride,
                                         int row0, int col0, int lane) {
  const int rlo = lane & 15, hi = lane >> 4;
  const bf16* p = base + (size_t)(row0 + rlo) * stride + col0 + 8 * hi;
  union { v16bf v; bf16x8 h[2]; } u;
  u.h[0] = *(const bf16x8*)(p);
  u.h[1] = *(const bf16x8*)(p + 16);
  return u.v;
}

__device__ __forceinline__ v8f wmma_bf16(v16bf a, v16bf b, v8f c) {
  return __builtin_amdgcn_wmma_f32_16x16x32_bf16(false, a, false, b,
                                                 (short)0, c, false, false);
}

// ---------------------------------------------------------------------------
// Kernel 1a: f32 -> bf16 pack (two elements per thread, one dword store)
// ---------------------------------------------------------------------------
__global__ void pack_bf16_pairs(const float* __restrict__ in,
                                unsigned* __restrict__ out, long long n2) {
  long long i = (long long)blockIdx.x * blockDim.x + threadIdx.x;
  if (i < n2) out[i] = pack2bf(in[2 * i], in[2 * i + 1]);
}

// ---------------------------------------------------------------------------
// Kernel 1b: transpose + pack: in (K x N) f32 row-major -> out (N x K) bf16.
// Makes WMMA B fragments contiguous-K (16B loads per lane).
// ---------------------------------------------------------------------------
__global__ void transpose_pack(const float* __restrict__ in,
                               bf16* __restrict__ out, int K, int N) {
  long long t = (long long)blockIdx.x * blockDim.x + threadIdx.x;
  if (t < (long long)K * N) {
    int k = (int)(t % K), n = (int)(t / K);
    out[(long long)n * K + k] = f2bf(in[(long long)k * N + n]);
  }
}

// ---------------------------------------------------------------------------
// Kernel 2: CPB bias. MLP 2->512(relu)->24 over 225 table rows, then gather
// to bias[h][n][m] (24 x 64 x 64 f32). Identical for every window.
// ---------------------------------------------------------------------------
__global__ void cpb_bias_kernel(const float* __restrict__ w1,
                                const float* __restrict__ b1,
                                const float* __restrict__ w2,
                                const float* __restrict__ rct,
                                const int*   __restrict__ rpi,
                                float* __restrict__ bias) {
  __shared__ float tbl[225 * 24];
  const int tid = threadIdx.x;
  if (tid < 225) {
    const float c0 = rct[2 * tid], c1 = rct[2 * tid + 1];
    float acc[24];
#pragma unroll
    for (int h = 0; h < 24; ++h) acc[h] = 0.f;
    for (int j = 0; j < 512; ++j) {
      float hv = fmaxf(c0 * w1[j] + c1 * w1[512 + j] + b1[j], 0.f);
#pragma unroll
      for (int h = 0; h < 24; ++h) acc[h] += hv * w2[j * 24 + h];
    }
#pragma unroll
    for (int h = 0; h < 24; ++h) tbl[tid * 24 + h] = acc[h];
  }
  __syncthreads();
  for (int p = tid; p < 24 * 4096; p += 256) {
    int h = p >> 12, nm = p & 4095;
    bias[p] = tbl[rpi[nm] * 24 + h];
  }
}

// ---------------------------------------------------------------------------
// Kernel 3: fused per-window attention. 1 block = 1 window, 8 waves, each
// wave owns 3 heads end-to-end (QKV gemm -> cosine norm -> S=qk^T+bias ->
// softmax (shfl_xor) -> P*V) using wave-level WMMA. x window staged into LDS
// via GLOBAL_LOAD_ASYNC_TO_LDS (ASYNCcnt) when available. Per-wave
// q/k/vT/scratch regions need no cross-wave sync (same-wave DS ops are
// in-order on CDNA5).
// ---------------------------------------------------------------------------
#define XSTR 776                     // 768 + 8 pad (keeps 16B alignment)
#define QSTR 40                      // q/k row stride (bf16)
#define VTSTR 72                     // v^T row stride (bf16)
#define SSTR 36                      // f32 scratch row stride
#define PSTR 72                      // P (bf16) row stride
#define XW_BYTES   (64 * XSTR * 2)                 // 99328
#define QK_BYTES   (64 * QSTR * 2)                 // 5120
#define VT_BYTES   (32 * VTSTR * 2)                // 4608
#define SCR_BYTES  9216                            // 64x36 f32 == 64x72 bf16
#define PW_BYTES   (2 * QK_BYTES + VT_BYTES + SCR_BYTES)   // 24064
#define ATTN_LDS   (XW_BYTES + 8 * PW_BYTES)       // 291840 < 320KB WGP LDS

__global__ __launch_bounds__(256) void attn_kernel(
    const bf16*  __restrict__ xb,       // (65536, 768) bf16
    const bf16*  __restrict__ wqkvT,    // (2304, 768) bf16  [col][k]
    const float* __restrict__ qkv_b,    // (2304,)
    const float* __restrict__ lscale,   // (24,)
    const float* __restrict__ bias,     // (24, 64, 64) f32
    bf16*        __restrict__ attn_out) // (65536, 768) bf16
{
  extern __shared__ char smem[];
  const int win  = blockIdx.x;
  const int tid  = threadIdx.x;
  const int wave = tid >> 5, lane = tid & 31;
  const int rlo  = lane & 15, hi = lane >> 4;

  bf16* xw = (bf16*)smem;
  char* wb = smem + XW_BYTES + wave * PW_BYTES;
  bf16*  qld  = (bf16*)wb;
  bf16*  kld  = (bf16*)(wb + QK_BYTES);
  bf16*  vT   = (bf16*)(wb + 2 * QK_BYTES);
  float* scrF = (float*)(wb + 2 * QK_BYTES + VT_BYTES);
  bf16*  scrP = (bf16*)(wb + 2 * QK_BYTES + VT_BYTES);

  // ---- stage the 64x768 bf16 window into LDS (16B chunks per lane) ----
  {
    const bf16* src = xb + (size_t)win * 64 * 768;
#if USE_ASYNC_LDS
    for (int t = tid; t < 64 * 96; t += 256) {
      int r = t / 96, c8 = (t % 96) * 8;
      async_copy16(src + r * 768 + c8, xw + r * XSTR + c8);
    }
    __builtin_amdgcn_s_wait_asynccnt(0);
#else
    for (int t = tid; t < 64 * 96; t += 256) {
      int r = t / 96, c8 = (t % 96) * 8;
      *(bf16x8*)(xw + r * XSTR + c8) = *(const bf16x8*)(src + r * 768 + c8);
    }
#endif
  }
  __syncthreads();

#pragma unroll 1
  for (int i = 0; i < 3; ++i) {
    const int   h   = wave * 3 + i;
    const float lsv = __expf(fminf(lscale[h], 4.60517018598809136f)); // ln(100)

    // -------------------- QKV: (64x768)@(768x32) per q/k/v --------------
#pragma unroll 1
    for (int which = 0; which < 3; ++which) {
      v8f acc[8];
#pragma unroll
      for (int t = 0; t < 8; ++t) acc[t] = vzero8();
      const bf16* wbase = wqkvT + (size_t)(which * 768 + h * 32) * 768;
#pragma unroll 1
      for (int ks = 0; ks < 24; ++ks) {
        const int kb = ks * 32;
        __builtin_prefetch(wbase + (size_t)rlo * 768 + kb + 32, 0, 0);
        v16bf a[4], b[2];
#pragma unroll
        for (int mt = 0; mt < 4; ++mt) a[mt] = frag_ld(xw, XSTR, 16 * mt, kb, lane);
#pragma unroll
        for (int nt = 0; nt < 2; ++nt) b[nt] = frag_ld(wbase, 768, 16 * nt, kb, lane);
#pragma unroll
        for (int mt = 0; mt < 4; ++mt)
#pragma unroll
          for (int nt = 0; nt < 2; ++nt)
            acc[mt * 2 + nt] = wmma_bf16(a[mt], b[nt], acc[mt * 2 + nt]);
      }
      float bq[2];
#pragma unroll
      for (int nt = 0; nt < 2; ++nt)
        bq[nt] = qkv_b[which * 768 + h * 32 + 16 * nt + rlo];

      if (which == 2) {
        // store V transposed: (m=token, n=d) -> vT[d][token]
#pragma unroll
        for (int mt = 0; mt < 4; ++mt)
#pragma unroll
          for (int nt = 0; nt < 2; ++nt)
#pragma unroll
            for (int r = 0; r < 8; ++r)
              vT[(16 * nt + rlo) * VTSTR + 16 * mt + r + 8 * hi] =
                  f2bf(acc[mt * 2 + nt][r] + bq[nt]);
      } else {
        // spill f32 to per-wave scratch, then per-lane row L2-normalize
#pragma unroll
        for (int mt = 0; mt < 4; ++mt)
#pragma unroll
          for (int nt = 0; nt < 2; ++nt)
#pragma unroll
            for (int r = 0; r < 8; ++r)
              scrF[(16 * mt + r + 8 * hi) * SSTR + 16 * nt + rlo] =
                  acc[mt * 2 + nt][r] + bq[nt];
        bf16* dst = (which == 0) ? qld : kld;
#pragma unroll
        for (int rr = 0; rr < 2; ++rr) {
          int row = lane * 2 + rr;
          float vals[32], ss = 0.f;
#pragma unroll
          for (int d = 0; d < 32; ++d) {
            float v = scrF[row * SSTR + d];
            vals[d] = v; ss += v * v;
          }
          float sc = ((which == 0) ? lsv : 1.0f) / fmaxf(sqrtf(ss), 1e-12f);
#pragma unroll
          for (int d = 0; d < 32; ++d) dst[row * QSTR + d] = f2bf(vals[d] * sc);
        }
      }
    }

    // -------------------- S = q @ k^T + bias (64x64, K=32) --------------
    v8f s[16];
#pragma unroll
    for (int t = 0; t < 16; ++t) s[t] = vzero8();
    {
      v16bf aq[4], bk[4];
#pragma unroll
      for (int mt = 0; mt < 4; ++mt) aq[mt] = frag_ld(qld, QSTR, 16 * mt, 0, lane);
#pragma unroll
      for (int nt = 0; nt < 4; ++nt) bk[nt] = frag_ld(kld, QSTR, 16 * nt, 0, lane);
#pragma unroll
      for (int mt = 0; mt < 4; ++mt)
#pragma unroll
        for (int nt = 0; nt < 4; ++nt)
          s[mt * 4 + nt] = wmma_bf16(aq[mt], bk[nt], s[mt * 4 + nt]);
    }
    const float* bh = bias + h * 4096;
#pragma unroll
    for (int mt = 0; mt < 4; ++mt)
#pragma unroll
      for (int nt = 0; nt < 4; ++nt)
#pragma unroll
        for (int r = 0; r < 8; ++r)
          s[mt * 4 + nt][r] += bh[(16 * mt + r + 8 * hi) * 64 + 16 * nt + rlo];

    // -------------------- softmax over n via shfl_xor within 16 lanes ---
#pragma unroll
    for (int mt = 0; mt < 4; ++mt) {
#pragma unroll
      for (int r = 0; r < 8; ++r) {
        float m = fmaxf(fmaxf(s[mt * 4 + 0][r], s[mt * 4 + 1][r]),
                        fmaxf(s[mt * 4 + 2][r], s[mt * 4 + 3][r]));
#pragma unroll
        for (int msk = 1; msk <= 8; msk <<= 1)
          m = fmaxf(m, __shfl_xor(m, msk, 32));
        float sum = 0.f;
#pragma unroll
        for (int nt = 0; nt < 4; ++nt) {
          float p = __expf(s[mt * 4 + nt][r] - m);
          s[mt * 4 + nt][r] = p;
          sum += p;
        }
#pragma unroll
        for (int msk = 1; msk <= 8; msk <<= 1)
          sum += __shfl_xor(sum, msk, 32);
        float inv = 1.0f / sum;
#pragma unroll
        for (int nt = 0; nt < 4; ++nt) s[mt * 4 + nt][r] *= inv;
      }
    }

    // round-trip P through LDS to convert D-layout -> A-layout
#pragma unroll
    for (int mt = 0; mt < 4; ++mt)
#pragma unroll
      for (int nt = 0; nt < 4; ++nt)
#pragma unroll
        for (int r = 0; r < 8; ++r)
          scrP[(16 * mt + r + 8 * hi) * PSTR + 16 * nt + rlo] =
              f2bf(s[mt * 4 + nt][r]);

    // -------------------- O = P @ V (64x32, K=64) -----------------------
    v8f o[8];
#pragma unroll
    for (int t = 0; t < 8; ++t) o[t] = vzero8();
#pragma unroll
    for (int ks = 0; ks < 2; ++ks) {
      v16bf ap[4], bv[2];
#pragma unroll
      for (int mt = 0; mt < 4; ++mt) ap[mt] = frag_ld(scrP, PSTR, 16 * mt, 32 * ks, lane);
#pragma unroll
      for (int nt = 0; nt < 2; ++nt) bv[nt] = frag_ld(vT, VTSTR, 16 * nt, 32 * ks, lane);
#pragma unroll
      for (int mt = 0; mt < 4; ++mt)
#pragma unroll
        for (int nt = 0; nt < 2; ++nt)
          o[mt * 2 + nt] = wmma_bf16(ap[mt], bv[nt], o[mt * 2 + nt]);
    }
    // write head output (bf16) into (B*N, C) with C index = h*32 + d
#pragma unroll
    for (int mt = 0; mt < 4; ++mt)
#pragma unroll
      for (int nt = 0; nt < 2; ++nt)
#pragma unroll
        for (int r = 0; r < 8; ++r) {
          int m = 16 * mt + r + 8 * hi, d = 16 * nt + rlo;
          attn_out[(size_t)(win * 64 + m) * 768 + h * 32 + d] =
              f2bf(o[mt * 2 + nt][r]);
        }
  }
}

// ---------------------------------------------------------------------------
// Kernel 4: output projection (65536x768)@(768x768)+bias -> f32 d_out.
// 128x128 block tile, 8 waves of 32x64. A chunk moved into LDS by the Tensor
// Data Mover: 2D tile 128x32 bf16, data_size=2B, LDS padding 16DW interval /
// 4DW amount -> DMA-produced 40-element row stride (bank-conflict free).
// B (transposed weights) streamed from L2 (1.2 MB, fully resident).
// ---------------------------------------------------------------------------
__global__ __launch_bounds__(256) void proj_kernel(
    const bf16*  __restrict__ A,   // (65536, 768) bf16
    const bf16*  __restrict__ BT,  // (768, 768) bf16  [col][k]
    const float* __restrict__ pb,  // (768,)
    float*       __restrict__ out) // (65536, 768) f32
{
  __shared__ alignas(16) bf16 As[128 * 40];
  const int tid = threadIdx.x, wave = tid >> 5, lane = tid & 31;
  const int rlo = lane & 15, hi = lane >> 4;
  const int m0 = blockIdx.y * 128, n0 = blockIdx.x * 128;
  const int wm = wave >> 1, wn = wave & 1;

  v8f acc[8];
#pragma unroll
  for (int t = 0; t < 8; ++t) acc[t] = vzero8();

#pragma unroll 1
  for (int ks = 0; ks < 24; ++ks) {
    const int kb = ks * 32;
    __syncthreads();
#if USE_TDM
    if (wave == 0) {
      // D# group0: count=1, lds_addr, 57b global byte address, type=2
      unsigned lds_off = (unsigned)(unsigned long long)(as3_void*)(void*)As;
      unsigned long long ga =
          (unsigned long long)(const void*)(A + (size_t)m0 * 768 + kb);
      u32x4 g0 = { 1u, lds_off, (unsigned)ga,
                   (unsigned)(ga >> 32) | (2u << 30) };
      // D# group1: data_size=2B, pad_enable, pad_interval=16DW, pad_amount=4DW
      // tensor_dim0=768, tensor_dim1=65536, tile_dim0=32, tile_dim1=128,
      // tensor_dim0_stride=768
      i32x8 g1 = { (1 << 16) | (1 << 20) | (3 << 22) | (3 << 25),
                   (int)(768u << 16),
                   0,
                   (int)(1u | (32u << 16)),
                   128,
                   768,
                   0,
                   0 };
      i32x4 gz = { 0, 0, 0, 0 };
#if __clang_major__ >= 23
      i32x8 g4 = { 0, 0, 0, 0, 0, 0, 0, 0 };
      __builtin_amdgcn_tensor_load_to_lds(g0, g1, gz, gz, g4, 0);
#else
      __builtin_amdgcn_tensor_load_to_lds(g0, g1, gz, gz, 0);
#endif
      __builtin_amdgcn_s_wait_tensorcnt(0);
    }
#else
    {
      int row = tid >> 1, c0 = (tid & 1) * 16;
      const bf16* src = A + (size_t)(m0 + row) * 768 + kb + c0;
      *(bf16x8*)(As + row * 40 + c0)     = *(const bf16x8*)(src);
      *(bf16x8*)(As + row * 40 + c0 + 8) = *(const bf16x8*)(src + 8);
      __builtin_prefetch(src + 32, 0, 0);  // next K chunk -> global_prefetch
    }
#endif
    __syncthreads();
    v16bf a[2], b[4];
#pragma unroll
    for (int mt = 0; mt < 2; ++mt)
      a[mt] = frag_ld(As, 40, wm * 32 + 16 * mt, 0, lane);
#pragma unroll
    for (int nt = 0; nt < 4; ++nt)
      b[nt] = frag_ld(BT, 768, n0 + wn * 64 + 16 * nt, kb, lane);
#pragma unroll
    for (int mt = 0; mt < 2; ++mt)
#pragma unroll
      for (int nt = 0; nt < 4; ++nt)
        acc[mt * 4 + nt] = wmma_bf16(a[mt], b[nt], acc[mt * 4 + nt]);
  }

  float pbv[4];
#pragma unroll
  for (int nt = 0; nt < 4; ++nt) pbv[nt] = pb[n0 + wn * 64 + 16 * nt + rlo];
#pragma unroll
  for (int mt = 0; mt < 2; ++mt)
#pragma unroll
    for (int nt = 0; nt < 4; ++nt)
#pragma unroll
      for (int r = 0; r < 8; ++r) {
        int m = m0 + wm * 32 + 16 * mt + r + 8 * hi;
        int n = n0 + wn * 64 + 16 * nt + rlo;
        out[(size_t)m * 768 + n] = acc[mt * 4 + nt][r] + pbv[nt];
      }
}

// ---------------------------------------------------------------------------
// Host launcher. ws layout (bytes):
//   xb       bf16 65536x768  : 100,663,296
//   wqkvT    bf16 2304x768   :   3,538,944
//   wprojT   bf16  768x768   :   1,179,648
//   bias     f32  24x64x64   :     393,216
//   attn_out bf16 65536x768  : 100,663,296   (total ~206.4 MB)
// ---------------------------------------------------------------------------
extern "C" void kernel_launch(void* const* d_in, const int* in_sizes, int n_in,
                              void* d_out, int out_size, void* d_ws, size_t ws_size,
                              hipStream_t stream) {
  (void)in_sizes; (void)n_in; (void)out_size; (void)ws_size;

  const float* x       = (const float*)d_in[0];
  const float* qkv_w   = (const float*)d_in[1];
  const float* qkv_b   = (const float*)d_in[2];
  const float* proj_w  = (const float*)d_in[3];
  const float* proj_b  = (const float*)d_in[4];
  const float* cpb_w1  = (const float*)d_in[5];
  const float* cpb_b1  = (const float*)d_in[6];
  const float* cpb_w2  = (const float*)d_in[7];
  const float* lscale  = (const float*)d_in[8];
  const float* rct     = (const float*)d_in[9];
  const int*   rpi     = (const int*)  d_in[10];
  float*       out     = (float*)d_out;

  char* ws = (char*)d_ws;
  size_t off = 0;
  bf16*  xb       = (bf16*)(ws + off); off += (size_t)65536 * 768 * 2;
  bf16*  wqkvT    = (bf16*)(ws + off); off += (size_t)2304 * 768 * 2;
  bf16*  wprojT   = (bf16*)(ws + off); off += (size_t)768 * 768 * 2;
  float* biasT    = (float*)(ws + off); off += (size_t)24 * 4096 * 4;
  bf16*  attn_out = (bf16*)(ws + off);

  const long long n2 = (long long)65536 * 768 / 2;
  pack_bf16_pairs<<<(unsigned)((n2 + 255) / 256), 256, 0, stream>>>(
      x, (unsigned*)xb, n2);
  transpose_pack<<<(768 * 2304 + 255) / 256, 256, 0, stream>>>(
      qkv_w, wqkvT, 768, 2304);
  transpose_pack<<<(768 * 768 + 255) / 256, 256, 0, stream>>>(
      proj_w, wprojT, 768, 768);
  cpb_bias_kernel<<<1, 256, 0, stream>>>(cpb_w1, cpb_b1, cpb_w2, rct, rpi, biasT);
  attn_kernel<<<1024, 256, ATTN_LDS, stream>>>(
      xb, wqkvT, qkv_b, lscale, biasT, attn_out);
  proj_kernel<<<dim3(6, 512), 256, 0, stream>>>(attn_out, wprojT, proj_b, out);
}
